// Gemma2Attention_88218628260579
// MI455X (gfx1250) — matrix operand order repaired and checked by reading, not compiled, not run on hardware
//
#include <hip/hip_runtime.h>
#include <hip/hip_bf16.h>

// ---------------------------------------------------------------------------
// Gemma2 attention layer for MI455X (gfx1250, wave32, WMMA, TDM + async-LDS)
//   S=4096, HIDDEN=2304, 8 Q heads / 4 KV heads, HEAD_DIM=256,
//   sliding_window=4096 (== S -> pure causal), softcap=50, scale=1/16
// ---------------------------------------------------------------------------

#define S_LEN   4096
#define HID     2304
#define NH      8
#define NKV     4
#define HD      256
#define QDIM    (NH * HD)    // 2048
#define KVDIM   (NKV * HD)   // 1024

typedef __attribute__((ext_vector_type(16))) __bf16 bf16x16;
typedef __attribute__((ext_vector_type(8)))  float  f32x8;
typedef int          v4i   __attribute__((vector_size(16)));
typedef unsigned int u32x4 __attribute__((vector_size(16)));
typedef int          i32x4 __attribute__((vector_size(16)));
typedef int          i32x8 __attribute__((vector_size(32)));

union FragAB {
    bf16x16 v;
    uint4   q[2];
};

__device__ __forceinline__ unsigned short f2bf(float f) {
    unsigned int u = __float_as_uint(f);
    if ((u & 0x7FFFFFFFu) > 0x7F800000u) return (unsigned short)0x7FC0u; // NaN
    unsigned int r = u + 0x7FFFu + ((u >> 16) & 1u); // round to nearest even
    return (unsigned short)(r >> 16);
}

// ---------------------------------------------------------------------------
// CDNA5 data movers: async global->LDS (ASYNCcnt) and TDM (TENSORcnt),
// both guarded with plain fallbacks.
// ---------------------------------------------------------------------------
#define AS1 __attribute__((address_space(1)))
#define AS3 __attribute__((address_space(3)))

#if __has_builtin(__builtin_amdgcn_global_load_async_to_lds_b128) && \
    __has_builtin(__builtin_amdgcn_s_wait_asynccnt)
#define HAVE_ASYNC_LDS 1
#else
#define HAVE_ASYNC_LDS 0
#endif

#if __has_builtin(__builtin_amdgcn_tensor_load_to_lds) && \
    __has_builtin(__builtin_amdgcn_s_wait_tensorcnt)
#define HAVE_TDM 1
#else
#define HAVE_TDM 0
#endif

__device__ __forceinline__ void copy16_g2l(const unsigned short* g, unsigned short* l) {
#if HAVE_ASYNC_LDS
    __builtin_amdgcn_global_load_async_to_lds_b128(
        (AS1 v4i*)(unsigned long long)g,
        (AS3 v4i*)(unsigned int)(unsigned long long)l, 0, 0);
#else
    *(uint4*)l = *(const uint4*)g;
#endif
}

__device__ __forceinline__ void async_wait0() {
#if HAVE_ASYNC_LDS
    __builtin_amdgcn_s_wait_asynccnt(0);
#endif
}

#if HAVE_TDM
// 2D tile DMA: tile_w elems (bf16) x tile_h rows, row stride in elems.
// LDS destination rows padded 64B data + 16B pad (matches LDT=40 ushorts):
// pad_interval=3 (16 DWORDs), pad_amount=3 (4 DWORDs).
__device__ __forceinline__ void tdm_load_tile_2d(const unsigned short* g,
                                                 const unsigned short* lds,
                                                 int tile_w, int tile_h,
                                                 int row_stride_elems) {
    unsigned long long ga = (unsigned long long)g;
    unsigned int       la = (unsigned int)(unsigned long long)lds;
    u32x4 g0;
    g0[0] = 1u;                                       // count=1, user descriptor
    g0[1] = la;                                       // lds_addr
    g0[2] = (unsigned int)ga;                         // global_addr[31:0]
    g0[3] = (unsigned int)((ga >> 32) & 0x01FFFFFFu)  // global_addr[56:32]
          | (2u << 30);                               // type=2 ("image")
    const int dim0 = 1 << 28, dim1 = 1 << 20;         // huge dims: no OOB clip
    i32x8 g1;
    g1[0] = (1 << 16) | (1 << 20) | (3 << 22) | (3 << 25); // 2B elems, pad enable
    g1[1] = (dim0 & 0xFFFF) << 16;                    // atomic_addr=0 | dim0 lo16
    g1[2] = (int)(((unsigned)dim0 >> 16) | (((unsigned)dim1 & 0xFFFFu) << 16));
    g1[3] = (int)((((unsigned)dim1 >> 16) & 0xFFFFu) | ((unsigned)tile_w << 16));
    g1[4] = tile_h;                                   // tile_dim1 | tile_dim2=0
    g1[5] = row_stride_elems;                         // tensor_dim0_stride lo32
    g1[6] = 0;                                        // stride hi | dim1_stride lo
    g1[7] = 0;
    i32x4 z4 = {0, 0, 0, 0};
    i32x8 z8 = {0, 0, 0, 0, 0, 0, 0, 0};
    __builtin_amdgcn_tensor_load_to_lds(g0, g1, z4, z4, z8, 0);
}
#endif

// ---------------------------------------------------------------------------
// f32 -> bf16 elementwise convert (vectorized x4)
// ---------------------------------------------------------------------------
__global__ void cvt_f32_bf16_kernel(const float* __restrict__ in,
                                    unsigned short* __restrict__ out, int n) {
    long i = ((long)blockIdx.x * blockDim.x + threadIdx.x) * 4;
    if (i >= n) return;
    float4 v = *(const float4*)(in + i);
    unsigned short o[4] = { f2bf(v.x), f2bf(v.y), f2bf(v.z), f2bf(v.w) };
    *(uint2*)(out + i) = *(uint2*)o;
}

// ---------------------------------------------------------------------------
// Weight transpose + f32 -> bf16:  WT[n][k] = bf16(W[k][n]).
// ---------------------------------------------------------------------------
__global__ void wtrans_bf16_kernel(const float* __restrict__ W,
                                   unsigned short* __restrict__ WT,
                                   int Kdim, int Ndim) {
    __shared__ float t[32][33];
    const int tx = threadIdx.x, ty = threadIdx.y;
    const int n0 = blockIdx.x * 32, k0 = blockIdx.y * 32;
#pragma unroll
    for (int j = 0; j < 4; ++j)
        t[ty + j * 8][tx] = W[(long)(k0 + ty + j * 8) * Ndim + n0 + tx];
    __syncthreads();
#pragma unroll
    for (int j = 0; j < 4; ++j)
        WT[(long)(n0 + ty + j * 8) * Kdim + k0 + tx] = f2bf(t[tx][ty + j * 8]);
}

// ---------------------------------------------------------------------------
// bf16 GEMM: C[M,N] = A[M,K](bf16 row-major) * BT[N,K](bf16 N-major)
// Block tile 128x128, 8 waves (2Mx4N), wave tile 64x32, K-step 32.
// Tile staging: TDM descriptors (one DMA per tile) with double buffering;
// falls back to per-lane async copies.
// ---------------------------------------------------------------------------
#define BM  128
#define BN  128
#define BK  32
#define LDT 40   // padded LDS row stride (ushorts): 80B, 16B-aligned

template <int K, int LDA, int LDBT, int LDC, bool TRANSB16>
__global__ __launch_bounds__(256)
void gemm_bf16_kernel(const unsigned short* __restrict__ A,
                      const unsigned short* __restrict__ BT,
                      void* __restrict__ Cv) {
    __shared__ __align__(16) unsigned short As[2][BM * LDT];
    __shared__ __align__(16) unsigned short Bs[2][BN * LDT];

    const int tid  = threadIdx.x;
    const int wave = tid >> 5;
    const int lane = tid & 31;
    const int lrow = lane & 15;
    const int hf   = lane >> 4;          // half-of-wave select
    const int wm   = (wave >> 2) * 64;   // wave M offset in tile
    const int wn   = (wave & 3) * 32;    // wave N offset in tile
    const int m0   = blockIdx.y * BM;
    const int n0   = blockIdx.x * BN;

    f32x8 acc[4][2];
#pragma unroll
    for (int ms = 0; ms < 4; ++ms)
#pragma unroll
        for (int ns = 0; ns < 2; ++ns)
#pragma unroll
            for (int r = 0; r < 8; ++r) acc[ms][ns][r] = 0.0f;

    constexpr int NK = K / BK;

#if HAVE_TDM
    auto stage = [&](int k0, int buf) {   // thread-0 only: 2 tile DMAs
        tdm_load_tile_2d(A  + m0 * LDA  + k0, &As[buf][0], BK, BM, LDA);
        tdm_load_tile_2d(BT + n0 * LDBT + k0, &Bs[buf][0], BK, BN, LDBT);
    };
    if (tid == 0) stage(0, 0);
#else
    const int srow = tid >> 2;
    const int scol = (tid & 3) * 8;
    auto stage = [&](int k0, int buf) {
#pragma unroll
        for (int i = 0; i < 2; ++i) {
            int row = srow + i * 64;
            copy16_g2l(A  + (m0 + row) * LDA  + k0 + scol, &As[buf][row * LDT + scol]);
            copy16_g2l(BT + (n0 + row) * LDBT + k0 + scol, &Bs[buf][row * LDT + scol]);
        }
    };
    stage(0, 0);
#endif

    for (int it = 0; it < NK; ++it) {
        const int buf = it & 1;
#if HAVE_TDM
        if (tid == 0) {
            if (it + 1 < NK) {
                stage((it + 1) * BK, buf ^ 1);            // prefetch next tile
                __builtin_amdgcn_s_wait_tensorcnt(2);     // current tile done
            } else {
                __builtin_amdgcn_s_wait_tensorcnt(0);
            }
        }
        __syncthreads();
#else
        async_wait0();
        __syncthreads();                 // tile `buf` complete, prev reads done
        if (it + 1 < NK) stage((it + 1) * BK, buf ^ 1);
#endif

        FragAB af[4], bfr[2];
#pragma unroll
        for (int ms = 0; ms < 4; ++ms) {
            int r = wm + ms * 16 + lrow;
            af[ms].q[0] = *(const uint4*)(&As[buf][r * LDT + hf * 8]);
            af[ms].q[1] = *(const uint4*)(&As[buf][r * LDT + 16 + hf * 8]);
        }
#pragma unroll
        for (int ns = 0; ns < 2; ++ns) {
            int nr = wn + ns * 16 + lrow;
            bfr[ns].q[0] = *(const uint4*)(&Bs[buf][nr * LDT + hf * 8]);
            bfr[ns].q[1] = *(const uint4*)(&Bs[buf][nr * LDT + 16 + hf * 8]);
        }
#pragma unroll
        for (int ms = 0; ms < 4; ++ms)
#pragma unroll
            for (int ns = 0; ns < 2; ++ns)
                acc[ms][ns] = __builtin_amdgcn_wmma_f32_16x16x32_bf16(
                    false, af[ms].v, false, bfr[ns].v,
                    (short)0, acc[ms][ns], false, false);

#if HAVE_TDM
        __syncthreads();   // all reads of `buf` done before it is re-staged
#endif
    }

#pragma unroll
    for (int ms = 0; ms < 4; ++ms)
#pragma unroll
        for (int ns = 0; ns < 2; ++ns)
#pragma unroll
            for (int r = 0; r < 8; ++r) {
                int row = m0 + wm + ms * 16 + r + 8 * hf;
                int col = n0 + wn + ns * 16 + lrow;
                if constexpr (TRANSB16) {
                    ((unsigned short*)Cv)[col * LDC + row] = f2bf(acc[ms][ns][r]);
                } else {
                    ((float*)Cv)[row * LDC + col] = acc[ms][ns][r];
                }
            }
}

// ---------------------------------------------------------------------------
// RoPE (rotate-half) + f32 -> bf16. One block = one (pos, head).
// ---------------------------------------------------------------------------
__global__ void rope_bf16_kernel(const float* __restrict__ X,
                                 unsigned short* __restrict__ Y,
                                 int nheads, int rowstride) {
    int b   = blockIdx.x;
    int pos = b / nheads;
    int h   = b - pos * nheads;
    int d   = threadIdx.x;            // 0..255
    int i   = d & 127;
    float inv = __expf(-(float)i * (9.210340371976184f / 128.0f)); // 10000^(-i/128)
    float ang = (float)pos * inv;
    float sn, cs;
    __sincosf(ang, &sn, &cs);
    const float* row = X + (long)pos * rowstride + h * HD;
    float x  = row[d];
    float xo = (d < 128) ? (x * cs - row[d + 128] * sn)
                         : (x * cs + row[d - 128] * sn);
    Y[(long)pos * rowstride + h * HD + d] = f2bf(xo);
}

// ---------------------------------------------------------------------------
// Flash attention: block = (qblock of 128, head). 8 waves, 16 queries/wave.
// Q/K row-major bf16 (fragments straight from global); V^T [KVDIM][S] bf16
// staged into double-buffered LDS tiles via async copies, overlapped with
// the score WMMAs. Softcap + causal + online softmax; 16x256 f32 ctx in regs.
// ---------------------------------------------------------------------------
__global__ __launch_bounds__(256)
void attn_kernel(const unsigned short* __restrict__ Qb,   // [S][2048] bf16
                 const unsigned short* __restrict__ Kb,   // [S][1024] bf16
                 const unsigned short* __restrict__ VbT,  // [1024][S] bf16
                 unsigned short* __restrict__ Ctx) {      // [S][2048] bf16
    __shared__ __align__(16) unsigned short Vt[2][HD * LDT];     // 2x 256x40
    __shared__ __align__(16) unsigned short Pl[8 * 16 * LDT];    // per-wave 16x40

    const int tid   = threadIdx.x;
    const int wave  = tid >> 5;
    const int lane  = tid & 31;
    const int lrow  = lane & 15;
    const int hf    = lane >> 4;
    const int h     = blockIdx.y;
    const int qblk  = blockIdx.x;
    const int q0    = qblk * 128 + wave * 16;
    const int qoff  = h * HD;
    const int kvoff = (h >> 1) * HD;

    // Q A-fragments for this wave's 16 rows: 8 k-steps of 32 features
    FragAB qf[8];
#pragma unroll
    for (int kk = 0; kk < 8; ++kk) {
        const unsigned short* p =
            Qb + (q0 + lrow) * QDIM + qoff + kk * 32 + hf * 8;
        qf[kk].q[0] = *(const uint4*)p;
        qf[kk].q[1] = *(const uint4*)(p + 16);
    }

    f32x8 ctx[16];
#pragma unroll
    for (int ft = 0; ft < 16; ++ft)
#pragma unroll
        for (int r = 0; r < 8; ++r) ctx[ft][r] = 0.0f;
    float rm[8], rl[8];
#pragma unroll
    for (int r = 0; r < 8; ++r) { rm[r] = -1e30f; rl[r] = 0.0f; }

    unsigned short* Pw = Pl + wave * 16 * LDT;
    const int nkb = (qblk + 1) * 4;   // 32-key blocks covering causal range
    const int vf0 = tid >> 2;         // V staging: feature row
    const int vkc = (tid & 3) * 8;    // V staging: key chunk

    auto stage_v = [&](int kb, int buf) {
#pragma unroll
        for (int i = 0; i < 4; ++i) {
            int f = vf0 + i * 64;
            copy16_g2l(VbT + (kvoff + f) * S_LEN + kb * 32 + vkc,
                       &Vt[buf][f * LDT + vkc]);
        }
    };

    stage_v(0, 0);

    for (int kb = 0; kb < nkb; ++kb) {
        const int buf  = kb & 1;
        const int key0 = kb * 32;
        async_wait0();
        __syncthreads();              // Vt[buf] ready; prev readers of buf^1 done
        if (kb + 1 < nkb) stage_v(kb + 1, buf ^ 1);   // overlap with scores

        // scores: two 16x16 tiles over 32 keys; K fragments from global
        f32x8 st[2];
#pragma unroll
        for (int t = 0; t < 2; ++t)
#pragma unroll
            for (int r = 0; r < 8; ++r) st[t][r] = 0.0f;
#pragma unroll
        for (int kk = 0; kk < 8; ++kk) {
#pragma unroll
            for (int t = 0; t < 2; ++t) {
                FragAB kf;
                const unsigned short* p = Kb +
                    (key0 + t * 16 + lrow) * KVDIM + kvoff + kk * 32 + hf * 8;
                kf.q[0] = *(const uint4*)p;
                kf.q[1] = *(const uint4*)(p + 16);
                st[t] = __builtin_amdgcn_wmma_f32_16x16x32_bf16(
                    false, qf[kk].v, false, kf.v, (short)0, st[t], false, false);
            }
        }

        // softcap + causal mask + online softmax (D-layout: row = r + 8*hf,
        // col = lrow; row-reduce across the 16-lane half via shfl_xor)
#pragma unroll
        for (int r = 0; r < 8; ++r) {
            int q  = q0 + r + 8 * hf;
            float mx = -1e30f;
#pragma unroll
            for (int t = 0; t < 2; ++t) {
                int key = key0 + t * 16 + lrow;
                float s = st[t][r] * 0.0625f;       // 1/sqrt(256)
                s = 50.0f * tanhf(s * 0.02f);       // softcap
                if (key > q) s = -1e30f;            // causal (window==S)
                st[t][r] = s;
                mx = fmaxf(mx, s);
            }
            mx = fmaxf(mx, __shfl_xor(mx, 1, 32));
            mx = fmaxf(mx, __shfl_xor(mx, 2, 32));
            mx = fmaxf(mx, __shfl_xor(mx, 4, 32));
            mx = fmaxf(mx, __shfl_xor(mx, 8, 32));
            float nm    = fmaxf(rm[r], mx);
            float alpha = __expf(rm[r] - nm);
            rm[r] = nm;
            float rs = 0.0f;
#pragma unroll
            for (int t = 0; t < 2; ++t) {
                float s = st[t][r];
                float p = (s < -1e29f) ? 0.0f : __expf(s - nm);
                rs += p;
                Pw[(r + 8 * hf) * LDT + t * 16 + lrow] = f2bf(p);
            }
            rs += __shfl_xor(rs, 1, 32);
            rs += __shfl_xor(rs, 2, 32);
            rs += __shfl_xor(rs, 4, 32);
            rs += __shfl_xor(rs, 8, 32);
            rl[r] = rl[r] * alpha + rs;
#pragma unroll
            for (int ft = 0; ft < 16; ++ft) ctx[ft][r] *= alpha;
        }

        // wave-local LDS fence for the P D->A-layout transpose bounce
        asm volatile("s_wait_dscnt 0" ::: "memory");

        FragAB pf;
        pf.q[0] = *(const uint4*)(&Pw[lrow * LDT + hf * 8]);
        pf.q[1] = *(const uint4*)(&Pw[lrow * LDT + 16 + hf * 8]);

#pragma unroll
        for (int ft = 0; ft < 16; ++ft) {
            FragAB vf;
            vf.q[0] = *(const uint4*)(&Vt[buf][(ft * 16 + lrow) * LDT + hf * 8]);
            vf.q[1] = *(const uint4*)(&Vt[buf][(ft * 16 + lrow) * LDT + 16 + hf * 8]);
            ctx[ft] = __builtin_amdgcn_wmma_f32_16x16x32_bf16(
                false, pf.v, false, vf.v, (short)0, ctx[ft], false, false);
        }
    }

    // epilogue: normalize and store bf16 context rows
#pragma unroll
    for (int r = 0; r < 8; ++r) {
        float inv = 1.0f / rl[r];
        int   row = (q0 + r + 8 * hf) * QDIM + qoff;
#pragma unroll
        for (int ft = 0; ft < 16; ++ft)
            Ctx[row + ft * 16 + lrow] = f2bf(ctx[ft][r] * inv);
    }
}

// ---------------------------------------------------------------------------
// Host-side orchestration
// ---------------------------------------------------------------------------
extern "C" void kernel_launch(void* const* d_in, const int* in_sizes, int n_in,
                              void* d_out, int out_size, void* d_ws, size_t ws_size,
                              hipStream_t stream) {
    const float* X  = (const float*)d_in[0];
    const float* Wq = (const float*)d_in[1];
    const float* Wk = (const float*)d_in[2];
    const float* Wv = (const float*)d_in[3];
    const float* Wo = (const float*)d_in[4];

    char* ws = (char*)d_ws;
    size_t o = 0;
    auto alloc_us = [&](size_t elems) {
        unsigned short* p = (unsigned short*)(ws + o);
        o = (o + elems * sizeof(unsigned short) + 255) & ~(size_t)255;
        return p;
    };
    auto alloc_f = [&](size_t elems) {
        float* p = (float*)(ws + o);
        o = (o + elems * sizeof(float) + 255) & ~(size_t)255;
        return p;
    };

    unsigned short* Xb   = alloc_us((size_t)S_LEN * HID);
    unsigned short* WqT  = alloc_us((size_t)QDIM * HID);    // [2048][2304]
    unsigned short* WkT  = alloc_us((size_t)KVDIM * HID);   // [1024][2304]
    unsigned short* WvT  = alloc_us((size_t)KVDIM * HID);
    unsigned short* WoT  = alloc_us((size_t)HID * QDIM);    // [2304][2048]
    float*          Qf   = alloc_f((size_t)S_LEN * QDIM);
    float*          Kf   = alloc_f((size_t)S_LEN * KVDIM);
    unsigned short* Qbf  = alloc_us((size_t)S_LEN * QDIM);
    unsigned short* Kbf  = alloc_us((size_t)S_LEN * KVDIM);
    unsigned short* VbT  = alloc_us((size_t)KVDIM * S_LEN); // [1024][4096]
    unsigned short* Cbf  = alloc_us((size_t)S_LEN * QDIM);

    // X -> bf16 (row-major); weights -> bf16 transposed (N-major)
    {
        long n = (long)S_LEN * HID;
        cvt_f32_bf16_kernel<<<(unsigned)((n / 4 + 255) / 256), 256, 0, stream>>>(X, Xb, (int)n);
    }
    wtrans_bf16_kernel<<<dim3(QDIM / 32,  HID / 32),  dim3(32, 8), 0, stream>>>(Wq, WqT, HID, QDIM);
    wtrans_bf16_kernel<<<dim3(KVDIM / 32, HID / 32),  dim3(32, 8), 0, stream>>>(Wk, WkT, HID, KVDIM);
    wtrans_bf16_kernel<<<dim3(KVDIM / 32, HID / 32),  dim3(32, 8), 0, stream>>>(Wv, WvT, HID, KVDIM);
    wtrans_bf16_kernel<<<dim3(HID / 32,   QDIM / 32), dim3(32, 8), 0, stream>>>(Wo, WoT, QDIM, HID);

    // Projections (Q,K -> f32 for RoPE; V -> bf16 transposed directly)
    gemm_bf16_kernel<HID, HID, HID, QDIM,  false>
        <<<dim3(QDIM / BN,  S_LEN / BM), 256, 0, stream>>>(Xb, WqT, Qf);
    gemm_bf16_kernel<HID, HID, HID, KVDIM, false>
        <<<dim3(KVDIM / BN, S_LEN / BM), 256, 0, stream>>>(Xb, WkT, Kf);
    gemm_bf16_kernel<HID, HID, HID, S_LEN, true>
        <<<dim3(KVDIM / BN, S_LEN / BM), 256, 0, stream>>>(Xb, WvT, VbT);

    // RoPE(Q,K) + bf16
    rope_bf16_kernel<<<S_LEN * NH,  256, 0, stream>>>(Qf, Qbf, NH,  QDIM);
    rope_bf16_kernel<<<S_LEN * NKV, 256, 0, stream>>>(Kf, Kbf, NKV, KVDIM);

    // Flash attention
    attn_kernel<<<dim3(S_LEN / 128, NH), 256, 0, stream>>>(Qbf, Kbf, VbT, Cbf);

    // Output projection -> f32 d_out
    gemm_bf16_kernel<QDIM, QDIM, QDIM, HID, false>
        <<<dim3(HID / BN, S_LEN / BM), 256, 0, stream>>>(Cbf, WoT, (float*)d_out);
}
